// FlashAttention_42915313221657
// MI455X (gfx1250) — matrix operand order repaired and checked by reading, not compiled, not run on hardware
//
#include <hip/hip_runtime.h>
#include <hip/hip_bf16.h>

typedef __attribute__((ext_vector_type(8)))  _Float16 v8h;
typedef __attribute__((ext_vector_type(16))) _Float16 v16h;
typedef __attribute__((ext_vector_type(8)))  float    v8f;

#define D_MODEL 512
#define N_HEADS 8
#define D_K     64
#define SEQ     2048
#define BQ      256   // query rows per workgroup
#define WQ      32    // query rows per wave (2 M-tiles of 16)
#define CK      32    // key/value rows per chunk
#define NW      8     // waves per workgroup (256 threads, wave32)

// LDS strides in halves (all multiples of 8 -> 16B-aligned v8h accesses)
#define SK_STRIDE 72
#define SV_STRIDE 40
#define SP_STRIDE 40

#define SHUF16(lo, hi) __builtin_shufflevector(lo, hi, 0,1,2,3,4,5,6,7,8,9,10,11,12,13,14,15)

// ---- DPP16 butterfly reductions within each 16-lane row (pure VALU, no LDS) ----
// ctrls: 0xB1 = quad_perm(1,0,3,2), 0x4E = quad_perm(2,3,0,1),
//        0x141 = row_half_mirror,   0x140 = row_mirror
__device__ __forceinline__ float rowmax16(float x) {
    union { float f; int i; } a, b;
    a.f = x;
    b.i = __builtin_amdgcn_update_dpp(0, a.i, 0xB1,  0xF, 0xF, true); a.f = fmaxf(a.f, b.f);
    b.i = __builtin_amdgcn_update_dpp(0, a.i, 0x4E,  0xF, 0xF, true); a.f = fmaxf(a.f, b.f);
    b.i = __builtin_amdgcn_update_dpp(0, a.i, 0x141, 0xF, 0xF, true); a.f = fmaxf(a.f, b.f);
    b.i = __builtin_amdgcn_update_dpp(0, a.i, 0x140, 0xF, 0xF, true); a.f = fmaxf(a.f, b.f);
    return a.f;
}
__device__ __forceinline__ float rowsum16(float x) {
    union { float f; int i; } a, b;
    a.f = x;
    b.i = __builtin_amdgcn_update_dpp(0, a.i, 0xB1,  0xF, 0xF, true); a.f += b.f;
    b.i = __builtin_amdgcn_update_dpp(0, a.i, 0x4E,  0xF, 0xF, true); a.f += b.f;
    b.i = __builtin_amdgcn_update_dpp(0, a.i, 0x141, 0xF, 0xF, true); a.f += b.f;
    b.i = __builtin_amdgcn_update_dpp(0, a.i, 0x140, 0xF, 0xF, true); a.f += b.f;
    return a.f;
}

__global__ __launch_bounds__(256)
void flash_attn_fwd_f16wmma(const float* __restrict__ Q,
                            const float* __restrict__ K,
                            const float* __restrict__ V,
                            float* __restrict__ O)
{
    __shared__ _Float16 sK[CK * SK_STRIDE];              // K chunk, row-major [row][d]
    __shared__ _Float16 sV[D_K * SV_STRIDE];             // V chunk, transposed [d][row]
    __shared__ _Float16 sP[NW * 2 * 16 * SP_STRIDE];     // per (wave, mtile) P staging

    const int tid    = threadIdx.x;
    const int lane   = tid & 31;
    const int wave   = tid >> 5;
    const int row16  = lane & 15;          // M / N index within 16
    const int hiHalf = (lane >> 4) & 1;    // 0: lanes 0-15, 1: lanes 16-31

    const int qtile = blockIdx.x;          // 0..7
    const int h     = blockIdx.y;          // 0..7
    const int b     = blockIdx.z;          // 0..3

    const float* Qbase = Q + ((size_t)b * SEQ) * D_MODEL + (size_t)h * D_K;
    const float* Kbase = K + ((size_t)b * SEQ) * D_MODEL + (size_t)h * D_K;
    const float* Vbase = V + ((size_t)b * SEQ) * D_MODEL + (size_t)h * D_K;
    float*       Obase = O + ((size_t)b * SEQ) * D_MODEL + (size_t)h * D_K;

    const int qbase = qtile * BQ + wave * WQ;

    // ---- Q fragments: [mtile][dchunk], each a 16x32 f16 A-frag.
    // Pre-scale by (1/sqrt(64)) * log2(e) so softmax runs in the exp2 domain.
    // A 16-bit 16x32 layout: lane<16 holds K={0..7,16..23}, lane>=16 holds K={8..15,24..31}.
    v16h aQ[2][2];
    {
        const float scale = 0.125f * 1.44269504088896340736f;
        const int off = hiHalf * 8;
        #pragma unroll
        for (int m = 0; m < 2; ++m) {
            const float* qrow = Qbase + (size_t)(qbase + m * 16 + row16) * D_MODEL;
            #pragma unroll
            for (int dc = 0; dc < 2; ++dc) {
                v16h a;
                #pragma unroll
                for (int j = 0; j < 8; ++j)
                    a[j]     = (_Float16)(qrow[dc * 32 + off + j]      * scale);
                #pragma unroll
                for (int j = 0; j < 8; ++j)
                    a[8 + j] = (_Float16)(qrow[dc * 32 + 16 + off + j] * scale);
                aQ[m][dc] = a;
            }
        }
    }

    // ---- accumulators: per mtile a 16x64 f32 output (4 C-tiles) + softmax row stats
    v8f acc[2][4];
    #pragma unroll
    for (int m = 0; m < 2; ++m)
        #pragma unroll
        for (int t = 0; t < 4; ++t) {
            v8f z = {0.f, 0.f, 0.f, 0.f, 0.f, 0.f, 0.f, 0.f};
            acc[m][t] = z;
        }
    float mrow[2][8], lrow[2][8];
    #pragma unroll
    for (int m = 0; m < 2; ++m)
        #pragma unroll
        for (int i = 0; i < 8; ++i) { mrow[m][i] = -__builtin_inff(); lrow[m][i] = 0.f; }

    const int nChunks = SEQ / CK;   // 64
    for (int c = 0; c < nChunks; ++c) {
        const int s0 = c * CK;

        // ---- cooperative fp32 -> f16 staging of K (row-major) and V (transposed)
        {
            const int r  = tid >> 3;           // 0..31
            const int c0 = (tid & 7) * 8;      // 0..56 step 8
            const float* kp = Kbase + (size_t)(s0 + r) * D_MODEL + c0;
            const float* vp = Vbase + (size_t)(s0 + r) * D_MODEL + c0;
            _Float16 kk[8], vv[8];
            #pragma unroll
            for (int j = 0; j < 8; ++j) { kk[j] = (_Float16)kp[j]; vv[j] = (_Float16)vp[j]; }
            v8h kvec;
            #pragma unroll
            for (int j = 0; j < 8; ++j) kvec[j] = kk[j];
            *(v8h*)&sK[r * SK_STRIDE + c0] = kvec;          // one ds_store_b128
            #pragma unroll
            for (int j = 0; j < 8; ++j) sV[(c0 + j) * SV_STRIDE + r] = vv[j];
            if (c + 1 < nChunks) {                           // global_prefetch_b8
                __builtin_prefetch(kp + (size_t)CK * D_MODEL, 0, 0);
                __builtin_prefetch(vp + (size_t)CK * D_MODEL, 0, 0);
            }
        }
        __syncthreads();

        // ---- S' = (Q * log2e/sqrt(d)) * K^T : 2 mtiles x 2 ktiles of 16x16 f32.
        // Each K B-fragment load feeds both mtiles (2x reuse per ds_load pair).
        v8f sc[2][2];
        #pragma unroll
        for (int m = 0; m < 2; ++m)
            #pragma unroll
            for (int kt = 0; kt < 2; ++kt) {
                v8f z = {0.f,0.f,0.f,0.f,0.f,0.f,0.f,0.f};
                sc[m][kt] = z;
            }
        #pragma unroll
        for (int dc = 0; dc < 2; ++dc) {
            const _Float16* kb0 = &sK[(0  + row16) * SK_STRIDE + dc * 32 + hiHalf * 8];
            const _Float16* kb1 = &sK[(16 + row16) * SK_STRIDE + dc * 32 + hiHalf * 8];
            v8h b0lo = *(const v8h*)kb0, b0hi = *(const v8h*)(kb0 + 16);
            v8h b1lo = *(const v8h*)kb1, b1hi = *(const v8h*)(kb1 + 16);
            v16h b0 = SHUF16(b0lo, b0hi);
            v16h b1 = SHUF16(b1lo, b1hi);
            #pragma unroll
            for (int m = 0; m < 2; ++m) {
                sc[m][0] = __builtin_amdgcn_wmma_f32_16x16x32_f16(false, aQ[m][dc], false, b0,
                                                                  (short)0, sc[m][0], false, false);
                sc[m][1] = __builtin_amdgcn_wmma_f32_16x16x32_f16(false, aQ[m][dc], false, b1,
                                                                  (short)0, sc[m][1], false, false);
            }
        }

        // ---- online softmax per mtile (DPP16 butterflies; exp2 domain)
        #pragma unroll
        for (int m = 0; m < 2; ++m) {
            _Float16* pw = &sP[(wave * 2 + m) * 16 * SP_STRIDE];
            float corr[8];
            #pragma unroll
            for (int i = 0; i < 8; ++i) {
                const float v    = rowmax16(fmaxf(sc[m][0][i], sc[m][1][i]));
                const float mnew = fmaxf(mrow[m][i], v);
                const float e0 = __builtin_amdgcn_exp2f(sc[m][0][i] - mnew);
                const float e1 = __builtin_amdgcn_exp2f(sc[m][1][i] - mnew);
                const float rs = rowsum16(e0 + e1);
                const float cf = __builtin_amdgcn_exp2f(mrow[m][i] - mnew);
                lrow[m][i] = lrow[m][i] * cf + rs;
                mrow[m][i] = mnew;
                corr[i] = cf;
                const int r = i + hiHalf * 8;
                pw[r * SP_STRIDE + row16]      = (_Float16)e0;
                pw[r * SP_STRIDE + 16 + row16] = (_Float16)e1;
            }
            #pragma unroll
            for (int t = 0; t < 4; ++t)
                #pragma unroll
                for (int i = 0; i < 8; ++i) acc[m][t][i] *= corr[i];
        }
        __syncthreads();   // cross-lane LDS visibility for P

        // ---- read P back as A-layout 16x32 fragments
        v16h pa[2];
        {
            const int off = hiHalf * 8;
            #pragma unroll
            for (int m = 0; m < 2; ++m) {
                const _Float16* pr = &sP[(wave * 2 + m) * 16 * SP_STRIDE + row16 * SP_STRIDE];
                v8h lo = *(const v8h*)(pr + off);
                v8h hi = *(const v8h*)(pr + 16 + off);
                pa[m] = SHUF16(lo, hi);
            }
        }

        // ---- O += P * V : each V B-fragment feeds both mtiles
        #pragma unroll
        for (int t = 0; t < 4; ++t) {
            const _Float16* vb = &sV[(t * 16 + row16) * SV_STRIDE + hiHalf * 8];
            v8h lo = *(const v8h*)vb, hi = *(const v8h*)(vb + 16);
            v16h bv = SHUF16(lo, hi);
            #pragma unroll
            for (int m = 0; m < 2; ++m)
                acc[m][t] = __builtin_amdgcn_wmma_f32_16x16x32_f16(false, pa[m], false, bv,
                                                                   (short)0, acc[m][t], false, false);
        }
        __syncthreads();   // before next chunk overwrites sK/sV
    }

    // ---- normalize and store (C layout: VGPR i -> row i + 8*hiHalf, col = 16t + row16)
    #pragma unroll
    for (int m = 0; m < 2; ++m)
        #pragma unroll
        for (int i = 0; i < 8; ++i) {
            const int r = i + hiHalf * 8;
            const float inv = 1.0f / lrow[m][i];
            float* orow = Obase + (size_t)(qbase + m * 16 + r) * D_MODEL;
            #pragma unroll
            for (int t = 0; t < 4; ++t)
                orow[t * 16 + row16] = acc[m][t][i] * inv;
        }
}

extern "C" void kernel_launch(void* const* d_in, const int* in_sizes, int n_in,
                              void* d_out, int out_size, void* d_ws, size_t ws_size,
                              hipStream_t stream) {
    (void)in_sizes; (void)n_in; (void)out_size; (void)d_ws; (void)ws_size;
    const float* Q = (const float*)d_in[0];
    const float* K = (const float*)d_in[1];
    const float* V = (const float*)d_in[2];
    float*       O = (float*)d_out;

    dim3 grid(SEQ / BQ, N_HEADS, 4 /*B*/);
    dim3 block(256);
    flash_attn_fwd_f16wmma<<<grid, block, 0, stream>>>(Q, K, V, O);
}
